// Policy_22033182228693
// MI455X (gfx1250) — matrix operand order repaired and checked by reading, not compiled, not run on hardware
//
#include <hip/hip_runtime.h>
#include <math.h>

typedef _Float16 v8h  __attribute__((ext_vector_type(8)));
typedef _Float16 v16h __attribute__((ext_vector_type(16)));
typedef float    v8f  __attribute__((ext_vector_type(8)));
typedef float    v4f  __attribute__((ext_vector_type(4)));

#define SEQ   40
#define H     128
#define ROWS  64          // batch rows per workgroup
#define MT    4           // 16-row m-tiles per workgroup
#define APAD  168         // padded A-stage row stride in f16 (160 used; 336B -> conflict-free b128)
#define KT    5           // K tiles of 32 over combined K = 160 ([w_rec | w_in | pad])

static __device__ __forceinline__ v8f v8f_zero() {
  v8f r;
#pragma unroll
  for (int i = 0; i < 8; ++i) r[i] = 0.0f;
  return r;
}

__global__ __launch_bounds__(256)
void snn_fused(const float* __restrict__ x,
               const float* __restrict__ w_in,
               const float* __restrict__ w_rec,
               const float* __restrict__ w_out,
               const float* __restrict__ mask,
               float* __restrict__ out, int Btot)
{
  __shared__ __align__(16) _Float16 Alds[ROWS * APAD];
  __shared__ float woutl[2 * H];

  const int tid  = threadIdx.x;
  const int lane = tid & 31;
  const int wv   = tid >> 5;       // wave id == n-tile: hidden cols [16*wv, 16*wv+16)
  const int l16  = lane & 15;
  const int sel  = lane >> 4;      // half-wave select
  const int b0   = blockIdx.x * ROWS;

  // ------------- persistent B fragments: split f32 -> f16 hi + f16 lo -------------
  // Combined B^T has K=160 rows: [w_rec k=0..127 | w_in k=128..135 | zeros].
  // B 32x16 fragment in this code's contiguous k order (same as A): lane col n = l16; lanes 0-15 hold K=kt*32+0..15,
  // lanes 16-31 hold K=kt*32+16..31 (2 f16 per VGPR, K ascending).
  v16h bh[KT], bl[KT];
  {
    const int n = 16 * wv + l16;
#pragma unroll
    for (int kt = 0; kt < KT; ++kt) {
      float vals[16];
      if (kt < 4) {
        const float* src = w_rec + n * H + 32 * kt + 16 * sel;
#pragma unroll
        for (int j = 0; j < 16; ++j) vals[j] = src[j];
      } else {
#pragma unroll
        for (int j = 0; j < 16; ++j) vals[j] = 0.0f;
        if (sel == 0) {
          const float* src = w_in + n * 8;
#pragma unroll
          for (int j = 0; j < 8; ++j) vals[j] = src[j];
        }
      }
#pragma unroll
      for (int j = 0; j < 16; ++j) {
        _Float16 hi = (_Float16)vals[j];
        bh[kt][j] = hi;
        bl[kt][j] = (_Float16)(vals[j] - (float)hi);
      }
    }
  }

  woutl[tid] = w_out[tid];                                  // 2*128 == blockDim
  for (int idx = tid; idx < ROWS * APAD; idx += 256) Alds[idx] = (_Float16)0.0f;

  // ------------- on-the-fly ConstantCurrentLIF encoder (2 channels / thread) ------
  const int er   = tid >> 2;        // row in tile, 0..63
  const int part = tid & 3;         // quarter of the hidden dim for readout
  const int f0   = 2 * part;        // features f0, f0+1
  float c0, c1;
  {
    const float s = (f0 < 4) ? 50.0f : -50.0f;
    c0 = fmaxf(0.0f, s * x[(b0 + er) * 4 + (f0 & 3)]);
    c1 = fmaxf(0.0f, s * x[(b0 + er) * 4 + ((f0 + 1) & 3)]);
  }
  float ve0 = 0.0f, ve1 = 0.0f;

  __syncthreads();
  { // encoder step -> x_0 into A-stage cols 128..135 (z region stays zero = z_{-1})
    ve0 = 0.9f * ve0 + 0.1f * c0;
    ve1 = 0.9f * ve1 + 0.1f * c1;
    float z0 = (ve0 > 1.0f) ? 1.0f : 0.0f;
    float z1 = (ve1 > 1.0f) ? 1.0f : 0.0f;
    ve0 *= (1.0f - z0); ve1 *= (1.0f - z1);
    Alds[er * APAD + H + f0]     = (_Float16)z0;
    Alds[er * APAD + H + f0 + 1] = (_Float16)z1;
  }
  __syncthreads();

  // hidden LIF state in WMMA accumulator layout: element r of m-tile m lives at
  // row 16*m + 8*sel + r, col 16*wv + l16
  v8f iacc[MT], vmem[MT];
#pragma unroll
  for (int m = 0; m < MT; ++m) { iacc[m] = v8f_zero(); vmem[m] = v8f_zero(); }

  float vo0 = 0.0f, io0 = 0.0f, mx0 = -1e30f;
  float vo1 = 0.0f, io1 = 0.0f, mx1 = -1e30f;

#pragma unroll 1
  for (int t = 0; t < SEQ; ++t) {
    // ---- LIF membrane update (uses i_t BEFORE synaptic input) ----
    v8h znh[MT];                                // spikes kept packed f16 (low VGPR cost)
#pragma unroll
    for (int m = 0; m < MT; ++m) {
#pragma unroll
      for (int r = 0; r < 8; ++r) {
        float vd = 0.9f * vmem[m][r] + 0.1f * iacc[m][r];
        float z  = (vd > 1.0f) ? 1.0f : 0.0f;
        vmem[m][r] = vd * (1.0f - z);
        znh[m][r]  = (_Float16)z;
      }
      iacc[m] = iacc[m] * 0.8f;     // synaptic decay
    }

    // ---- i += [z_{t-1} | x_t] @ [w_rec | w_in]^T  (split-f16 WMMA, fp32-exact) ----
    // A fragment: lanes 0-15 take K [0..15] of the 32-K tile,
    // lanes 16-31 take [16..31]; row M = 16*m + l16.
#pragma unroll
    for (int kt = 0; kt < KT; ++kt) {
#pragma unroll
      for (int m = 0; m < MT; ++m) {
        const _Float16* ap = &Alds[(16 * m + l16) * APAD + 32 * kt + 16 * sel];
        v8h p0 = *(const v8h*)ap;
        v8h p1 = *(const v8h*)(ap + 8);
        v16h a = __builtin_shufflevector(p0, p1,
                   0,1,2,3,4,5,6,7,8,9,10,11,12,13,14,15);
        iacc[m] = __builtin_amdgcn_wmma_f32_16x16x32_f16(
            false, a, false, bh[kt], (short)0, iacc[m], false, false);
        iacc[m] = __builtin_amdgcn_wmma_f32_16x16x32_f16(
            false, a, false, bl[kt], (short)0, iacc[m], false, false);
      }
    }

    __syncthreads();   // every wave finished reading the A stage

    // ---- publish z_t; encoder emits x_{t+1} ----
#pragma unroll
    for (int m = 0; m < MT; ++m)
#pragma unroll
      for (int r = 0; r < 8; ++r)
        Alds[(16 * m + 8 * sel + r) * APAD + 16 * wv + l16] = znh[m][r];

    {
      ve0 = 0.9f * ve0 + 0.1f * c0;
      ve1 = 0.9f * ve1 + 0.1f * c1;
      float z0 = (ve0 > 1.0f) ? 1.0f : 0.0f;
      float z1 = (ve1 > 1.0f) ? 1.0f : 0.0f;
      ve0 *= (1.0f - z0); ve1 *= (1.0f - z1);
      Alds[er * APAD + H + f0]     = (_Float16)z0;
      Alds[er * APAD + H + f0 + 1] = (_Float16)z1;
    }

    __syncthreads();   // z_t visible to readers

    // ---- readout: io += (z_t * mask_t) @ w_out^T ; track running max of vo ----
    // mask is a 335MB stream read exactly once (> 192MB L2): use non-temporal loads
    const float*    mrow = mask + ((size_t)t * Btot + (b0 + er)) * (size_t)H + 32 * part;
    const _Float16* zrow = &Alds[er * APAD + 32 * part];
    float s0 = 0.0f, s1 = 0.0f;
#pragma unroll
    for (int j = 0; j < 32; j += 4) {
      v4f mk = __builtin_nontemporal_load((const v4f*)(mrow + j));
      float zd0 = (float)zrow[j + 0] * mk.x;
      float zd1 = (float)zrow[j + 1] * mk.y;
      float zd2 = (float)zrow[j + 2] * mk.z;
      float zd3 = (float)zrow[j + 3] * mk.w;
      const int h = 32 * part + j;
      s0 += zd0 * woutl[h]   + zd1 * woutl[h+1]   + zd2 * woutl[h+2]   + zd3 * woutl[h+3];
      s1 += zd0 * woutl[H+h] + zd1 * woutl[H+h+1] + zd2 * woutl[H+h+2] + zd3 * woutl[H+h+3];
    }
    if (t + 1 < SEQ) __builtin_prefetch((const void*)(mrow + (size_t)Btot * H), 0, 1);

    s0 += __shfl_xor(s0, 1, 32); s0 += __shfl_xor(s0, 2, 32);
    s1 += __shfl_xor(s1, 1, 32); s1 += __shfl_xor(s1, 2, 32);

    float von0 = 0.9f * vo0 + 0.1f * io0;   // LI readout uses OLD io
    float von1 = 0.9f * vo1 + 0.1f * io1;
    io0 = 0.8f * io0 + s0;
    io1 = 0.8f * io1 + s1;
    mx0 = fmaxf(mx0, von0); vo0 = von0;
    mx1 = fmaxf(mx1, von1); vo1 = von1;
  }

  if (part == 0) {
    float mmx = fmaxf(mx0, mx1);
    float e0 = expf(mx0 - mmx), e1 = expf(mx1 - mmx);
    float inv = 1.0f / (e0 + e1);
    out[(b0 + er) * 2 + 0] = e0 * inv;
    out[(b0 + er) * 2 + 1] = e1 * inv;
  }
}

extern "C" void kernel_launch(void* const* d_in, const int* in_sizes, int n_in,
                              void* d_out, int out_size, void* d_ws, size_t ws_size,
                              hipStream_t stream) {
  (void)n_in; (void)out_size; (void)d_ws; (void)ws_size;
  const float* x     = (const float*)d_in[0];
  const float* w_in  = (const float*)d_in[1];
  const float* w_rec = (const float*)d_in[2];
  const float* w_out = (const float*)d_in[3];
  const float* mask  = (const float*)d_in[4];
  float*       outp  = (float*)d_out;

  const int Btot = in_sizes[0] / 4;          // 16384
  dim3 grid(Btot / ROWS), block(256);
  hipLaunchKernelGGL(snn_fused, grid, block, 0, stream,
                     x, w_in, w_rec, w_out, mask, outp, Btot);
}
